// ScalableDAG_v1_32993938768553
// MI455X (gfx1250) — matrix operand order repaired and checked by reading, not compiled, not run on hardware
//
#include <hip/hip_runtime.h>

typedef __attribute__((ext_vector_type(16))) _Float16 v16h;
typedef __attribute__((ext_vector_type(8)))  float    v8f;

#define N_BATCH 16384
#define D_DIM   128
#define M1_DIM  64
#define K_DIM   8

#define B_ROW_PITCH 136   // 128 halves + 8 pad: 68 dwords/row -> bank = 4*m + c, conflict-free

union V16Pack { v16h v; uint4 q[2]; };

// ---------- prologue kernels: precompute f16 operands + fused tail weights ----------

__global__ __launch_bounds__(256) void prep_x_f16(const float* __restrict__ x,
                                                  _Float16* __restrict__ xh, int n) {
  int i = blockIdx.x * 256 + threadIdx.x;
  if (i < n) xh[i] = (_Float16)x[i];
}

__global__ __launch_bounds__(256) void prep_w_f16(const float* __restrict__ wp,
                                                  const float* __restrict__ wn,
                                                  _Float16* __restrict__ wh, int n) {
  int i = blockIdx.x * 256 + threadIdx.x;
  if (i < n) {
    int j = i >> 13;           // i / (M1*D)
    int d = i & (D_DIM - 1);   // fastest dim of [D, M1, D]
    float v = wp[i] - wn[i];
    if (d == j) v = 0.0f;      // diagonal mask: node j ignores x[:, j]
    wh[i] = (_Float16)v;
  }
}

__global__ __launch_bounds__(256) void prep_tail(const float* __restrict__ b1p,
                                                 const float* __restrict__ b1n,
                                                 const float* __restrict__ w2,
                                                 const float* __restrict__ b2,
                                                 const float* __restrict__ w3,
                                                 float* __restrict__ bb,
                                                 float* __restrict__ vwt,
                                                 float* __restrict__ cvec) {
  int i = blockIdx.x * 256 + threadIdx.x;
  if (i < D_DIM * M1_DIM) {
    int j = i >> 6, m = i & (M1_DIM - 1);
    bb[i] = b1p[i] - b1n[i];
    float a = 0.0f;
#pragma unroll
    for (int k = 0; k < K_DIM; ++k) a += w3[j * K_DIM + k] * w2[k * M1_DIM + m];
    vwt[i] = a;
    if (m == 0) {
      float c = 0.0f;
#pragma unroll
      for (int k = 0; k < K_DIM; ++k) c += w3[j * K_DIM + k] * b2[k];
      cvec[j] = c;
    }
  }
}

// ---------- main fused kernel ----------
// Block = one node j, 8 waves = 8 consecutive 16-row n-tiles.
// B panel for node j (64x128 f16 = 16KB) staged once into LDS, shared by all waves.
// Inner loop software-pipelined: fragment t+1 loads issue before the WMMA on t.

__global__ __launch_bounds__(256) void dag_wmma_main(const _Float16* __restrict__ Xh,
                                                     const _Float16* __restrict__ Wh,
                                                     const float* __restrict__ bb,
                                                     const float* __restrict__ vwt,
                                                     const float* __restrict__ cvec,
                                                     float* __restrict__ out) {
  __shared__ _Float16 Bsh[M1_DIM * B_ROW_PITCH];   // 64 * 136 * 2 = 17,408 B

  const int lane   = threadIdx.x & 31;
  const int waveid = threadIdx.x >> 5;
  const int j      = blockIdx.x & (D_DIM - 1);     // adjacent blocks share X rows in L2
  const int ngrp   = blockIdx.x >> 7;
  const int ntile  = ngrp * 8 + waveid;
  const int half   = lane >> 4;                    // lane half: K sub-range / C row half
  const int row    = lane & 15;                    // A row (n) / B column (m) / C column (m)
  const int nbase  = ntile << 4;

  // ---- cooperative fill: Wh[j] (64 rows x 128 halves) -> LDS with padded pitch ----
  {
    const _Float16* WhJ = Wh + ((size_t)j << 13);  // j * 64 * 128
#pragma unroll
    for (int q = threadIdx.x; q < 1024; q += 256) {
      const int m   = q >> 4;        // row 0..63
      const int seg = q & 15;        // 16 x 8-half segments per row
      uint4 v = *(const uint4*)(WhJ + m * D_DIM + seg * 8);
      *(uint4*)(&Bsh[m * B_ROW_PITCH + seg * 8]) = v;
    }
  }

  const _Float16* Xrow = Xh + (size_t)(nbase + row) * D_DIM;
  // lane's base within a B row: K-range select (lanes0-15: K=0..15, lanes16-31: K=16..31)
  const _Float16* Brow = &Bsh[row * B_ROW_PITCH + 16 * half];

  // Accumulators: bias folded into C init (bias is constant down each column).
  v8f   acc[4];
  float vv[4];
#pragma unroll
  for (int t = 0; t < 4; ++t) {
    const int m = t * 16 + row;
    const float b = bb[j * M1_DIM + m];
    vv[t] = vwt[j * M1_DIM + m];
#pragma unroll
    for (int r = 0; r < 8; ++r) acc[t][r] = b;
  }

  __syncthreads();

  // D=128 contraction: 4 K-steps of 32. A from global (L2), B from LDS.
  // Rotation pipeline: only 2 A buffers + 2 B buffers live; each load issues
  // one WMMA ahead of its consumer so waits keep one request in flight.
  V16Pack A0, A1, B0, B1;
  // prologue loads: A(s=0), B(t=0,s=0)
  A0.q[0] = *(const uint4*)(Xrow + 16 * half);
  A0.q[1] = *(const uint4*)(Xrow + 16 * half + 8);
  B0.q[0] = *(const uint4*)(Brow);
  B0.q[1] = *(const uint4*)(Brow + 8);

#pragma unroll
  for (int s = 0; s < 4; ++s) {
    if (s < 3) {  // prefetch next K-step's A fragment
      A1.q[0] = *(const uint4*)(Xrow + 32 * (s + 1) + 16 * half);
      A1.q[1] = *(const uint4*)(Xrow + 32 * (s + 1) + 16 * half + 8);
    }
#pragma unroll
    for (int t = 0; t < 4; ++t) {
      if (t < 3) {        // prefetch next m-tile's B fragment (same K-step)
        const _Float16* bp = Brow + (t + 1) * 16 * B_ROW_PITCH + 32 * s;
        B1.q[0] = *(const uint4*)(bp);
        B1.q[1] = *(const uint4*)(bp + 8);
      } else if (s < 3) { // prefetch first B fragment of next K-step
        const _Float16* bp = Brow + 32 * (s + 1);
        B1.q[0] = *(const uint4*)(bp);
        B1.q[1] = *(const uint4*)(bp + 8);
      }
      acc[t] = __builtin_amdgcn_wmma_f32_16x16x32_f16(
          /*neg_a=*/false, A0.v, /*neg_b=*/false, B0.v,
          /*c_mod=*/(short)0, acc[t], /*reuse_a=*/false, /*reuse_b=*/false);
      B0 = B1;   // rename after full unroll
    }
    A0 = A1;     // rename after full unroll
  }

  // Fused tail: out[n,j] = sum_m sigmoid(h) * v[j,m] + c[j]
  float sum[8];
#pragma unroll
  for (int r = 0; r < 8; ++r) {
    float srtot = 0.0f;
#pragma unroll
    for (int t = 0; t < 4; ++t) {
      float sig = 1.0f / (1.0f + __expf(-acc[t][r]));
      srtot += sig * vv[t];
    }
    sum[r] = srtot;
  }

  // Butterfly reduce 8 row-sums across the 16-lane column group; final lane l
  // holds the total of sum[l&7] over its 16-lane half.
  float t4[4];
#pragma unroll
  for (int r = 0; r < 4; ++r) {
    float lo = sum[2 * r]     + __shfl_xor(sum[2 * r],     1, 32);
    float hi = sum[2 * r + 1] + __shfl_xor(sum[2 * r + 1], 1, 32);
    t4[r] = (lane & 1) ? hi : lo;
  }
  float t2[2];
#pragma unroll
  for (int r = 0; r < 2; ++r) {
    float lo = t4[2 * r]     + __shfl_xor(t4[2 * r],     2, 32);
    float hi = t4[2 * r + 1] + __shfl_xor(t4[2 * r + 1], 2, 32);
    t2[r] = (lane & 2) ? hi : lo;
  }
  float lo  = t2[0] + __shfl_xor(t2[0], 4, 32);
  float hi  = t2[1] + __shfl_xor(t2[1], 4, 32);
  float red = (lane & 4) ? hi : lo;
  red += __shfl_xor(red, 8, 32);

  const float cj = cvec[j];
  if ((lane & 8) == 0) {
    const int n = nbase + half * 8 + (lane & 7);   // lanes16-31 hold rows 8..15
    out[(size_t)n * D_DIM + j] = red + cj;
  }
}

// ---------- launcher ----------

extern "C" void kernel_launch(void* const* d_in, const int* in_sizes, int n_in,
                              void* d_out, int out_size, void* d_ws, size_t ws_size,
                              hipStream_t stream) {
  const float* x   = (const float*)d_in[0];
  const float* w1p = (const float*)d_in[1];
  const float* b1p = (const float*)d_in[2];
  const float* w1n = (const float*)d_in[3];
  const float* b1n = (const float*)d_in[4];
  const float* w2  = (const float*)d_in[5];
  const float* b2  = (const float*)d_in[6];
  const float* w3  = (const float*)d_in[7];
  float* out = (float*)d_out;

  char* ws = (char*)d_ws;
  _Float16* Xh   = (_Float16*)(ws);                 // 16384*128*2  = 4,194,304 B
  _Float16* Wh   = (_Float16*)(ws + 4194304);       // 128*64*128*2 = 2,097,152 B
  float*    bb   = (float*)   (ws + 6291456);       // 8192*4 = 32,768 B
  float*    vwt  = (float*)   (ws + 6324224);       // 8192*4 = 32,768 B
  float*    cvec = (float*)   (ws + 6356992);       // 128*4  = 512 B

  const int nX = N_BATCH * D_DIM;                   // 2,097,152
  const int nW = D_DIM * M1_DIM * D_DIM;            // 1,048,576

  prep_x_f16<<<nX / 256, 256, 0, stream>>>(x, Xh, nX);
  prep_w_f16<<<nW / 256, 256, 0, stream>>>(w1p, w1n, Wh, nW);
  prep_tail <<<(D_DIM * M1_DIM + 255) / 256, 256, 0, stream>>>(b1p, b1n, w2, b2, w3,
                                                               bb, vwt, cvec);

  // blocks = (N/16 n-tiles / 8 per block) * D nodes = 128 * 128 = 16384
  const int blocks = (N_BATCH / 128) * D_DIM;
  dag_wmma_main<<<blocks, 256, 0, stream>>>(Xh, Wh, bb, vwt, cvec, out);
}
